// DeepseekMoEBlock_84232898609560
// MI455X (gfx1250) — matrix operand order repaired and checked by reading, not compiled, not run on hardware
//
#include <hip/hip_runtime.h>

// Problem constants (match reference)
#define T_TOK 1024
#define DIM   2048
#define DFF_  1024
#define NE    16
#define TOPK  6

typedef __attribute__((ext_vector_type(16))) __bf16 v16bf;
typedef __attribute__((ext_vector_type(8)))  __bf16 v8bf;
typedef __attribute__((ext_vector_type(8)))  float  v8f;

// Build a v16bf WMMA fragment from two 16-byte LDS chunks.
static __device__ __forceinline__ v16bf ld_frag2(const __bf16* p0, const __bf16* p1) {
    union { v16bf v; v8bf h[2]; } u;
    u.h[0] = *(const v8bf*)p0;
    u.h[1] = *(const v8bf*)p1;
    return u.v;
}

// Pack 8 f32 (two float4) into one 16B bf16 vector for a single ds_store_b128.
static __device__ __forceinline__ v8bf cvt8(float4 lo, float4 hi) {
    v8bf r;
    r[0] = (__bf16)lo.x; r[1] = (__bf16)lo.y; r[2] = (__bf16)lo.z; r[3] = (__bf16)lo.w;
    r[4] = (__bf16)hi.x; r[5] = (__bf16)hi.y; r[6] = (__bf16)hi.z; r[7] = (__bf16)hi.w;
    return r;
}

// Fast silu: g * rcp(1+exp(-g)) -- v_exp_f32 + v_rcp_f32, no IEEE div chain.
static __device__ __forceinline__ float fast_silu(float g) {
    return g * __builtin_amdgcn_rcpf(1.0f + __expf(-g));
}

// ---------------------------------------------------------------------------
// Kernel 1: routing. Compact per-expert token lists + merged top-k weights.
// ---------------------------------------------------------------------------
__global__ void moe_route(const int* __restrict__ sel, const float* __restrict__ rw,
                          int* __restrict__ cnt, int* __restrict__ tok,
                          float* __restrict__ tw) {
    int t = blockIdx.x * blockDim.x + threadIdx.x;
    if (t >= T_TOK) return;
    int   e[TOPK];
    float w[TOPK];
#pragma unroll
    for (int k = 0; k < TOPK; ++k) { e[k] = sel[t * TOPK + k]; w[k] = rw[t * TOPK + k]; }
#pragma unroll
    for (int k = 0; k < TOPK; ++k) {
        bool first = true;
#pragma unroll
        for (int j = 0; j < TOPK; ++j) if (j < k && e[j] == e[k]) first = false;
        if (!first) continue;                       // expert already pushed by earlier slot
        float s = w[k];
#pragma unroll
        for (int j = 0; j < TOPK; ++j) if (j > k && e[j] == e[k]) s += w[j];
        int ee   = e[k] & (NE - 1);
        int slot = atomicAdd(&cnt[ee], 1);
        tok[ee * T_TOK + slot] = t;
        tw [ee * T_TOK + slot] = s;
    }
}

// ---------------------------------------------------------------------------
// Kernel 2: gate+up GEMMs (bf16 WMMA, f32 acc) + SwiGLU -> h (bf16) workspace.
// Block: 256 threads = 8 wave32 (2 x 4 wave grid). Tile: 64 tokens x 128 DFF.
// Software pipeline: next K-slice global loads issue before current compute.
// ---------------------------------------------------------------------------
__global__ __launch_bounds__(256)
void moe_gateup(const float* __restrict__ x,  const float* __restrict__ w0,
                const float* __restrict__ w1, const float* __restrict__ s0,
                const float* __restrict__ s1,
                const int* __restrict__ cnt,  const int* __restrict__ tok,
                __bf16* __restrict__ hws) {
    const int e      = blockIdx.z;
    const int n_base = blockIdx.y * 128;
    const int m_base = blockIdx.x * 64;
    const int count  = cnt[e];
    if (m_base >= count) return;                    // uniform per block

    __shared__ __align__(16) __bf16 As [64][40];    // 80B row stride: 16B aligned, bank-spread
    __shared__ __align__(16) __bf16 B0s[128][40];
    __shared__ __align__(16) __bf16 B1s[128][40];

    const int tid  = threadIdx.x;
    const int wave = tid >> 5, lane = tid & 31;
    const int wm   = wave >> 2, wn  = wave & 3;     // 2x4 wave grid
    const int lane16 = lane & 15, hsel = lane >> 4;

    // Staging maps: x tile 64x32 (8 f32/thread), w tiles 128x32 (16 f32/thread each)
    const int arow = tid >> 2, acol = (tid & 3) * 8;
    int slotA = m_base + arow; if (slotA >= count) slotA = m_base;   // clamp pad rows
    const int tA = tok[e * T_TOK + slotA];
    const float* xsrc  = x  + (size_t)tA * DIM + acol;
    const int brow = tid >> 1, bcol = (tid & 1) * 16;
    const float* w0src = w0 + ((size_t)e * DFF_ + n_base + brow) * DIM + bcol;
    const float* w1src = w1 + ((size_t)e * DFF_ + n_base + brow) * DIM + bcol;

    float4 pa[2], pb0[4], pb1[4];                   // prefetch registers
#define LOAD_PF_A(K0)                                                       \
    do {                                                                    \
        pa[0]  = *(const float4*)(xsrc  + (K0));                            \
        pa[1]  = *(const float4*)(xsrc  + (K0) + 4);                        \
        _Pragma("unroll")                                                   \
        for (int i = 0; i < 4; ++i) {                                       \
            pb0[i] = *(const float4*)(w0src + (K0) + 4 * i);                \
            pb1[i] = *(const float4*)(w1src + (K0) + 4 * i);                \
        }                                                                   \
    } while (0)

    v8f accG[2][2] = {}; v8f accU[2][2] = {};

    LOAD_PF_A(0);
    for (int k0 = 0; k0 < DIM; k0 += 32) {
        __syncthreads();                            // previous compute done: LDS writable
        *(v8bf*)&As [arow][acol]     = cvt8(pa[0],  pa[1]);
        *(v8bf*)&B0s[brow][bcol]     = cvt8(pb0[0], pb0[1]);
        *(v8bf*)&B0s[brow][bcol + 8] = cvt8(pb0[2], pb0[3]);
        *(v8bf*)&B1s[brow][bcol]     = cvt8(pb1[0], pb1[1]);
        *(v8bf*)&B1s[brow][bcol + 8] = cvt8(pb1[2], pb1[3]);
        if (k0 + 32 < DIM) LOAD_PF_A(k0 + 32);      // overlap next loads with compute
        __syncthreads();                            // LDS ready

        v16bf aF[2], b0F[2], b1F[2];
#pragma unroll
        for (int sm = 0; sm < 2; ++sm) {            // A: K contiguous, halves by lane group
            const __bf16* r = &As[wm * 32 + sm * 16 + lane16][0];
            aF[sm] = ld_frag2(r + hsel * 16, r + hsel * 16 + 8);
        }
#pragma unroll
        for (int sn = 0; sn < 2; ++sn) {            // B: K contiguous, halves by lane group
            const __bf16* r0 = &B0s[wn * 32 + sn * 16 + lane16][0];
            const __bf16* r1 = &B1s[wn * 32 + sn * 16 + lane16][0];
            b0F[sn] = ld_frag2(r0 + hsel * 16, r0 + hsel * 16 + 8);
            b1F[sn] = ld_frag2(r1 + hsel * 16, r1 + hsel * 16 + 8);
        }
#pragma unroll
        for (int sm = 0; sm < 2; ++sm)
#pragma unroll
            for (int sn = 0; sn < 2; ++sn) {
                accG[sm][sn] = __builtin_amdgcn_wmma_f32_16x16x32_bf16(
                    false, aF[sm], false, b0F[sn], (short)0, accG[sm][sn], false, false);
                accU[sm][sn] = __builtin_amdgcn_wmma_f32_16x16x32_bf16(
                    false, aF[sm], false, b1F[sn], (short)0, accU[sm][sn], false, false);
            }
    }
#undef LOAD_PF_A

    // SwiGLU epilogue: h = silu(g*s0) * (u*s1), stored bf16
    const float sc0 = s0[e], sc1 = s1[e];
#pragma unroll
    for (int sm = 0; sm < 2; ++sm)
#pragma unroll
        for (int v = 0; v < 8; ++v) {
            const int mloc = wm * 32 + sm * 16 + hsel * 8 + v;
            if (m_base + mloc >= count) continue;
            __bf16* hrow = hws + ((size_t)(e * T_TOK + m_base + mloc)) * DFF_ + n_base;
#pragma unroll
            for (int sn = 0; sn < 2; ++sn) {
                float g = accG[sm][sn][v] * sc0;
                float u = accU[sm][sn][v] * sc1;
                hrow[wn * 32 + sn * 16 + lane16] = (__bf16)(fast_silu(g) * u);
            }
        }
}

// ---------------------------------------------------------------------------
// Kernel 3: down GEMM y = h @ w2^T, scaled by s2 * cw, scatter-add into out.
// ---------------------------------------------------------------------------
__global__ __launch_bounds__(256)
void moe_down(const float* __restrict__ w2, const float* __restrict__ s2,
              const int* __restrict__ cnt, const int* __restrict__ tok,
              const float* __restrict__ tw, const __bf16* __restrict__ hws,
              float* __restrict__ out) {
    const int e      = blockIdx.z;
    const int n_base = blockIdx.y * 128;            // over DIM
    const int m_base = blockIdx.x * 64;
    const int count  = cnt[e];
    if (m_base >= count) return;

    __shared__ __align__(16) __bf16 As[64][40];
    __shared__ __align__(16) __bf16 Bs[128][40];

    const int tid  = threadIdx.x;
    const int wave = tid >> 5, lane = tid & 31;
    const int wm   = wave >> 2, wn  = wave & 3;
    const int lane16 = lane & 15, hsel = lane >> 4;

    const int arow = tid >> 2, acol = (tid & 3) * 8;
    int slotA = m_base + arow; if (slotA >= count) slotA = m_base;
    const __bf16* hsrc = hws + ((size_t)(e * T_TOK) + slotA) * DFF_ + acol;
    const int brow = tid >> 1, bcol = (tid & 1) * 16;
    const float* w2src = w2 + ((size_t)e * DIM + n_base + brow) * DFF_ + bcol;

    v8bf   pa;                                       // h already bf16: direct 16B load
    float4 pb[4];
#define LOAD_PF_B(K0)                                                       \
    do {                                                                    \
        pa = *(const v8bf*)(hsrc + (K0));                                   \
        _Pragma("unroll")                                                   \
        for (int i = 0; i < 4; ++i)                                         \
            pb[i] = *(const float4*)(w2src + (K0) + 4 * i);                 \
    } while (0)

    v8f acc[2][2] = {};

    LOAD_PF_B(0);
    for (int k0 = 0; k0 < DFF_; k0 += 32) {
        __syncthreads();
        *(v8bf*)&As[arow][acol]     = pa;
        *(v8bf*)&Bs[brow][bcol]     = cvt8(pb[0], pb[1]);
        *(v8bf*)&Bs[brow][bcol + 8] = cvt8(pb[2], pb[3]);
        if (k0 + 32 < DFF_) LOAD_PF_B(k0 + 32);
        __syncthreads();

        v16bf aF[2], bF[2];
#pragma unroll
        for (int sm = 0; sm < 2; ++sm) {
            const __bf16* r = &As[wm * 32 + sm * 16 + lane16][0];
            aF[sm] = ld_frag2(r + hsel * 16, r + hsel * 16 + 8);
        }
#pragma unroll
        for (int sn = 0; sn < 2; ++sn) {
            const __bf16* r = &Bs[wn * 32 + sn * 16 + lane16][0];
            bF[sn] = ld_frag2(r + hsel * 16, r + hsel * 16 + 8);
        }
#pragma unroll
        for (int sm = 0; sm < 2; ++sm)
#pragma unroll
            for (int sn = 0; sn < 2; ++sn)
                acc[sm][sn] = __builtin_amdgcn_wmma_f32_16x16x32_bf16(
                    false, aF[sm], false, bF[sn], (short)0, acc[sm][sn], false, false);
    }
#undef LOAD_PF_B

    const float sc2 = s2[e];
#pragma unroll
    for (int sm = 0; sm < 2; ++sm)
#pragma unroll
        for (int v = 0; v < 8; ++v) {
            const int mloc = wm * 32 + sm * 16 + hsel * 8 + v;
            const int slot = m_base + mloc;
            if (slot >= count) continue;
            const int   t   = tok[e * T_TOK + slot];
            const float wcw = tw [e * T_TOK + slot] * sc2;
            float* orow = out + (size_t)t * DIM + n_base;
#pragma unroll
            for (int sn = 0; sn < 2; ++sn)
                atomicAdd(&orow[wn * 32 + sn * 16 + lane16], acc[sm][sn][v] * wcw);
        }
}

// ---------------------------------------------------------------------------
// Launcher
// ---------------------------------------------------------------------------
extern "C" void kernel_launch(void* const* d_in, const int* in_sizes, int n_in,
                              void* d_out, int out_size, void* d_ws, size_t ws_size,
                              hipStream_t stream) {
    const float* x   = (const float*)d_in[0];
    const float* w0  = (const float*)d_in[1];
    const float* w1  = (const float*)d_in[2];
    const float* w2  = (const float*)d_in[3];
    const float* s0  = (const float*)d_in[4];
    const float* s1  = (const float*)d_in[5];
    const float* s2  = (const float*)d_in[6];
    const int*   sel = (const int*)d_in[7];
    const float* rw  = (const float*)d_in[8];
    float*       out = (float*)d_out;

    // Workspace layout: [cnt: 256B][tok: E*T ints][tw: E*T floats][h: E*T*DFF bf16]
    char*   ws  = (char*)d_ws;
    int*    cnt = (int*)ws;
    int*    tok = (int*)(ws + 256);
    float*  tw  = (float*)(ws + 256 + (size_t)NE * T_TOK * 4);
    __bf16* hws = (__bf16*)(ws + 256 + 2 * (size_t)NE * T_TOK * 4);

    hipMemsetAsync(cnt, 0, 256, stream);
    hipMemsetAsync(out, 0, (size_t)T_TOK * DIM * sizeof(float), stream);

    moe_route<<<(T_TOK + 255) / 256, 256, 0, stream>>>(sel, rw, cnt, tok, tw);
    moe_gateup<<<dim3(T_TOK / 64, DFF_ / 128, NE), 256, 0, stream>>>(
        x, w0, w1, s0, s1, cnt, tok, hws);
    moe_down<<<dim3(T_TOK / 64, DIM / 128, NE), 256, 0, stream>>>(
        w2, s2, cnt, tok, tw, hws, out);
}
